// GTDCALayer_4063039062084
// MI455X (gfx1250) — matrix operand order repaired and checked by reading, not compiled, not run on hardware
//
#include <hip/hip_runtime.h>

typedef __attribute__((ext_vector_type(16))) _Float16 v16h;
typedef __attribute__((ext_vector_type(8)))  _Float16 v8h;
typedef __attribute__((ext_vector_type(8)))  float    v8f;
typedef __attribute__((ext_vector_type(2)))  float    v2f;
typedef __attribute__((ext_vector_type(4)))  unsigned int u32x4;
typedef __attribute__((ext_vector_type(8)))  int      i32x8;
typedef __attribute__((ext_vector_type(4)))  int      i32x4;

#define DINL __device__ __forceinline__

#if defined(__has_builtin)
#  if __has_builtin(__builtin_amdgcn_tensor_load_to_lds) && \
      __has_builtin(__builtin_amdgcn_s_wait_tensorcnt)
#    define HAVE_TDM 1
#  else
#    define HAVE_TDM 0
#  endif
#else
#  define HAVE_TDM 0
#endif

// ---------------------------------------------------------------------------
// WMMA fragment helpers (CDNA5 wave32 layouts, CDNA5 ISA §7.12.2)
// A 16x32 f16: lane row = lane&15; elem e<8 -> K=k0+e+(lane>=16?8:0),
//                                  e>=8   -> K=k0+16+(e-8)+(lane>=16?8:0)
// B 32x16 f16: lane col = lane&15; same K mapping as A
// C/D 16x16 f32: elem i -> M = i + (lane>=16?8:0), N = lane&15
// ---------------------------------------------------------------------------
DINL v16h load_a16(const _Float16* __restrict__ A, int lda, int row0, int k0, int lane) {
    const _Float16* p = A + (size_t)(row0 + (lane & 15)) * lda + k0 + ((lane >> 4) << 3);
    v16h a;
#pragma unroll
    for (int e = 0; e < 8; ++e) { a[e] = p[e]; a[8 + e] = p[16 + e]; }
    return a;
}

DINL v16h load_b16(const _Float16* __restrict__ W, int ldw, int n0, int k0, int lane) {
    const _Float16* p = W + (size_t)(n0 + (lane & 15)) * ldw + k0 + ((lane >> 4) << 3);
    v16h b;
#pragma unroll
    for (int e = 0; e < 8; ++e) { b[e] = p[e]; b[8 + e] = p[16 + e]; }
    return b;
}

DINL v8f wmma16(v16h a, v16h b, v8f c) {
    return __builtin_amdgcn_wmma_f32_16x16x32_f16(false, a, false, b, (short)0, c,
                                                  false, false);
}

// ---------------------------------------------------------------------------
// TDM: 1-D DMA of `nelem` f16 elements from global to LDS (contiguous).
// D# per CDNA5 ISA:
//   group0: [1:0]=count=1, [63:32]=lds_addr, [120:64]=global_addr, [127:126]=2
//   group1: data_size=1 (2B) @[17:16]; tensor_dim0 @[79:48]; tensor_dim1 @[111:80];
//           tile_dim0 @[127:112]; tile_dim1 @[143:128]; dim0_stride @[207:160];
//           dim1_stride @[255:208]
// ---------------------------------------------------------------------------
#if HAVE_TDM
DINL void tdm_load_lds_2b(unsigned lds_off, const void* gsrc, unsigned nelem) {
    unsigned long long ga = (unsigned long long)gsrc;
    u32x4 g0;
    g0[0] = 1u;                                   // count=1, user mode, no gather
    g0[1] = lds_off;                              // lds_addr (bytes)
    g0[2] = (unsigned)ga;                         // global_addr[31:0]
    g0[3] = (unsigned)(ga >> 32) | 0x80000000u;   // global_addr[56:32] | type=2<<30
    i32x8 g1;
    g1[0] = 1 << 16;                              // data_size = 2 bytes
    g1[1] = (int)((nelem & 0xFFFFu) << 16);       // tensor_dim0[15:0]
    g1[2] = (int)(((nelem >> 16) & 0xFFFFu) | (1u << 16)); // dim0[31:16] | tensor_dim1=1
    g1[3] = (int)((nelem & 0xFFFFu) << 16);       // tile_dim0 (= nelem, <= 65535)
    g1[4] = 1;                                    // tile_dim1=1, tile_dim2=0
    g1[5] = (int)nelem;                           // tensor_dim0_stride[31:0]
    g1[6] = (int)((nelem & 0xFFFFu) << 16);       // dim0_stride[47:32]=0 | dim1_stride[15:0]
    g1[7] = 0;                                    // dim1_stride[47:16]
    i32x4 z = {0, 0, 0, 0};
#if __clang_major__ >= 23
    i32x8 z8 = {0, 0, 0, 0, 0, 0, 0, 0};
    __builtin_amdgcn_tensor_load_to_lds(g0, g1, z, z, z8, 0);
#else
    __builtin_amdgcn_tensor_load_to_lds(g0, g1, z, z, 0);
#endif
}
#endif

// ---------------------------------------------------------------------------
// Epilogue functors
// ---------------------------------------------------------------------------
struct EpiF32 {                      // plain row-major f32
    float* out; int ldo;
    __device__ void operator()(int r, int c, float v) const {
        out[(size_t)r * ldo + c] = v;
    }
};
struct EpiQ {                        // q -> (b, m, q, ch) f16
    _Float16* out;
    __device__ void operator()(int r, int c, float v) const {
        int b = r >> 13, q = r & 8191, m = c >> 5, ch = c & 31;
        out[(((size_t)(b * 8 + m) * 8192) + q) * 32 + ch] = (_Float16)v;
    }
};
struct EpiK {                        // k -> (b, m, t, ch) f16
    _Float16* out;
    __device__ void operator()(int r, int c, float v) const {
        int b = r >> 9, t = r & 511, m = c >> 5, ch = c & 31;
        out[(((size_t)(b * 8 + m) * 512) + t) * 32 + ch] = (_Float16)v;
    }
};
struct EpiV {                        // v -> (b, m, ch, t) f16 (transposed for PV GEMM)
    _Float16* out;
    __device__ void operator()(int r, int c, float v) const {
        int b = r >> 9, t = r & 511, m = c >> 5, ch = c & 31;
        out[(((size_t)(b * 8 + m) * 32) + ch) * 512 + t] = (_Float16)v;
    }
};
struct EpiAttw {                     // fused softmax over groups of 4 adjacent cols (P=4)
    float* out;                      // (32768, 32)
    __device__ void operator()(int r, int c, float v) const {
        float mx = v;
        mx = fmaxf(mx, __shfl_xor(mx, 1));
        mx = fmaxf(mx, __shfl_xor(mx, 2));
        float e = __expf(v - mx);
        float s = e;
        s += __shfl_xor(s, 1);
        s += __shfl_xor(s, 2);
        out[(size_t)r * 32 + c] = e / s;
    }
};

// ---------------------------------------------------------------------------
// WMMA f16 GEMM-NT: out[m,n] = sum_k A[m,k]*W[n,k] + bias[n].  All-f16 inputs.
// Block = 4 waves / one 64-row x 64-col macro-tile.  The 64xKD weight tile
// (contiguous 32 KB) is DMA'd to LDS once per block via the Tensor Data Mover
// (fallback: cooperative vector copy), then all 4 waves read B-fragments from
// LDS while streaming their A rows from global.
// ---------------------------------------------------------------------------
template <int NTILES, int KD, typename Epi>
__global__ __launch_bounds__(128) void gemm_nt_f16k(const _Float16* __restrict__ A,
                                                    const _Float16* __restrict__ W,
                                                    const float* __restrict__ bias,
                                                    Epi epi) {
    __shared__ __attribute__((aligned(16))) _Float16 Wlds[NTILES * 16 * KD];
    int lane = threadIdx.x & 31;
    int wave = threadIdx.x >> 5;
    int row0 = (blockIdx.x * 4 + wave) * 16;
    int col0 = blockIdx.y * (NTILES * 16);
    const _Float16* wtile = W + (size_t)col0 * KD;

#if HAVE_TDM
    if (wave == 0) {
        unsigned lds_off = (unsigned)(unsigned long long)(void*)&Wlds[0];
        tdm_load_lds_2b(lds_off, wtile, NTILES * 16 * KD);
        __builtin_amdgcn_s_wait_tensorcnt(0);
    }
#else
    {
        const uint4* src = (const uint4*)wtile;
        uint4* dst = (uint4*)&Wlds[0];
#pragma unroll
        for (int i = threadIdx.x; i < NTILES * 16 * KD / 8; i += 128) dst[i] = src[i];
    }
#endif
    __syncthreads();

    v8f acc[NTILES];
#pragma unroll
    for (int t = 0; t < NTILES; ++t)
#pragma unroll
        for (int i = 0; i < 8; ++i) acc[t][i] = 0.0f;

    for (int k0 = 0; k0 < KD; k0 += 32) {
        v16h a = load_a16(A, KD, row0, k0, lane);
#pragma unroll
        for (int t = 0; t < NTILES; ++t) {
            v16h b = load_b16(Wlds, KD, t * 16, k0, lane);   // ds_load path
            acc[t] = wmma16(a, b, acc[t]);
        }
    }
    int rbase = row0 + ((lane >> 4) << 3);
#pragma unroll
    for (int t = 0; t < NTILES; ++t) {
        int c = col0 + t * 16 + (lane & 15);
        float bv = bias[c];
#pragma unroll
        for (int i = 0; i < 8; ++i) epi(rbase + i, c, acc[t][i] + bv);
    }
}

// ---------------------------------------------------------------------------
// WMMA f32 GEMM-NT (V_WMMA_F32_16X16X4_F32): full-precision path for the
// coordinate-critical offset / attention-weight heads.
// A 16x4 frag: elem e -> K = k0 + e + (lane>=16?2:0); B symmetric.
// ---------------------------------------------------------------------------
template <int NTILES, typename Epi>
__global__ __launch_bounds__(128) void gemm_nt_f32k(const float* __restrict__ A,
                                                    const float* __restrict__ W,
                                                    const float* __restrict__ bias,
                                                    int K_, Epi epi) {
    int lane = threadIdx.x & 31;
    int wave = threadIdx.x >> 5;
    int row0 = (blockIdx.x * 4 + wave) * 16;
    int col0 = blockIdx.y * (NTILES * 16);

    v8f acc[NTILES];
#pragma unroll
    for (int t = 0; t < NTILES; ++t)
#pragma unroll
        for (int i = 0; i < 8; ++i) acc[t][i] = 0.0f;

    int rA = row0 + (lane & 15);
    int koff = (lane >> 4) << 1;
    for (int k0 = 0; k0 < K_; k0 += 4) {
        const float* pa = A + (size_t)rA * K_ + k0 + koff;
        v2f a; a[0] = pa[0]; a[1] = pa[1];
#pragma unroll
        for (int t = 0; t < NTILES; ++t) {
            const float* pw = W + (size_t)(col0 + t * 16 + (lane & 15)) * K_ + k0 + koff;
            v2f b; b[0] = pw[0]; b[1] = pw[1];
            acc[t] = __builtin_amdgcn_wmma_f32_16x16x4_f32(false, a, false, b, (short)0,
                                                           acc[t], false, false);
        }
    }
    int rbase = row0 + ((lane >> 4) << 3);
#pragma unroll
    for (int t = 0; t < NTILES; ++t) {
        int c = col0 + t * 16 + (lane & 15);
        float bv = bias[c];
#pragma unroll
        for (int i = 0; i < 8; ++i) epi(rbase + i, c, acc[t][i] + bv);
    }
}

// ---------------------------------------------------------------------------
// f32 -> f16 conversion (weights / activations pre-pass)
// ---------------------------------------------------------------------------
__global__ __launch_bounds__(256) void cvt_kernel(const float* __restrict__ in,
                                                  _Float16* __restrict__ out, int n) {
    int idx = blockIdx.x * 256 + threadIdx.x;
    if (idx < n) out[idx] = (_Float16)in[idx];
}

// ---------------------------------------------------------------------------
// Trajectory projection: tf[n,c] = tp[n,0]*w[c,0] + tp[n,1]*w[c,1] + b[c] (f16 out)
// ---------------------------------------------------------------------------
__global__ __launch_bounds__(256) void traj_kernel(const float* __restrict__ tp,
                                                   const float* __restrict__ w,
                                                   const float* __restrict__ b,
                                                   _Float16* __restrict__ tf) {
    int idx = blockIdx.x * 256 + threadIdx.x;   // B*Nt*C = 524288
    int c = idx & 255;
    int n = idx >> 8;
    tf[idx] = (_Float16)(tp[n * 2] * w[c * 2] + tp[n * 2 + 1] * w[c * 2 + 1] + b[c]);
}

// ---------------------------------------------------------------------------
// Fused per-head attention: softmax(q k^T / sqrt(32)) v
// One wave = one (b, m, 16-row q-tile).  Scores streamed through a per-wave
// transposed LDS buffer S[t][row] (one v8h store per tile), two-pass softmax.
// ---------------------------------------------------------------------------
__global__ __launch_bounds__(64) void attn_kernel(const _Float16* __restrict__ qf,
                                                  const _Float16* __restrict__ kf,
                                                  const _Float16* __restrict__ vf,
                                                  _Float16* __restrict__ ctx) {
    __shared__ v8h S[2][512][2];        // [wave][t][rowhalf] : 16 KB per wave
    __shared__ float rstat[2][16];

    int lane = threadIdx.x & 31;
    int wave = threadIdx.x >> 5;
    int gid  = blockIdx.x * 2 + wave;     // 16384 tiles total
    int qt = gid & 511;
    int bm = gid >> 9;                    // b*8+m
    int b = bm >> 3, m = bm & 7;
    int q0 = qt * 16;

    const _Float16* qbase = qf + (size_t)bm * 8192 * 32;
    const _Float16* kbase = kf + (size_t)bm * 512 * 32;
    const _Float16* vbase = vf + (size_t)bm * 32 * 512;

    v16h aq = load_a16(qbase, 32, q0, 0, lane);   // K = Ch = 32 exactly

    const float scale = 0.17677669529663687f;     // 1/sqrt(32)
    int rhalf = (lane >> 4) << 3;                 // C-frag row offset (0 or 8)
    float vmax[8];
#pragma unroll
    for (int i = 0; i < 8; ++i) vmax[i] = -1e30f;

    // Pass 1: scores -> LDS (transposed), track row max
    for (int tt = 0; tt < 32; ++tt) {
        v16h kb = load_b16(kbase, 32, tt * 16, 0, lane);
        v8f s = {0.f, 0.f, 0.f, 0.f, 0.f, 0.f, 0.f, 0.f};
        s = wmma16(aq, kb, s);
        int tcol = tt * 16 + (lane & 15);
        v8h hs;
#pragma unroll
        for (int i = 0; i < 8; ++i) {
            float sv = s[i] * scale;
            vmax[i] = fmaxf(vmax[i], sv);
            hs[i] = (_Float16)sv;
        }
        S[wave][tcol][lane >> 4] = hs;            // 16B ds_store per lane
    }
#pragma unroll
    for (int mask = 1; mask < 16; mask <<= 1)
#pragma unroll
        for (int i = 0; i < 8; ++i) vmax[i] = fmaxf(vmax[i], __shfl_xor(vmax[i], mask));
    if ((lane & 15) == 0) {
#pragma unroll
        for (int i = 0; i < 8; ++i) rstat[wave][rhalf + i] = vmax[i];
    }
    __syncthreads();

    int myrow = lane & 15;
    float rmax = rstat[wave][myrow];
    __syncthreads();

    // Pass 2: P = exp(S - max); ctx = P @ V (two 16-col N tiles = 32 channels)
    v8f acc0 = {0.f, 0.f, 0.f, 0.f, 0.f, 0.f, 0.f, 0.f};
    v8f acc1 = {0.f, 0.f, 0.f, 0.f, 0.f, 0.f, 0.f, 0.f};
    float psum = 0.f;
    const _Float16* sbase = (const _Float16*)&S[wave][0][0];  // (t,row): t*16 + row
    for (int kt = 0; kt < 16; ++kt) {
        int k0 = kt * 32;
        v16h p;
#pragma unroll
        for (int e = 0; e < 8; ++e) {
            float sv = (float)sbase[(k0 + rhalf + e) * 16 + myrow];
            float pe = __expf(sv - rmax);
            psum += pe;
            p[e] = (_Float16)pe;
            float sv2 = (float)sbase[(k0 + 16 + rhalf + e) * 16 + myrow];
            float pe2 = __expf(sv2 - rmax);
            psum += pe2;
            p[8 + e] = (_Float16)pe2;
        }
        v16h b0 = load_b16(vbase, 512, 0, k0, lane);
        v16h b1 = load_b16(vbase, 512, 16, k0, lane);
        acc0 = wmma16(p, b0, acc0);
        acc1 = wmma16(p, b1, acc1);
    }
    psum += __shfl_xor(psum, 16);
    if (lane < 16) rstat[wave][lane] = 1.0f / psum;
    __syncthreads();

#pragma unroll
    for (int i = 0; i < 8; ++i) {
        int r = rhalf + i;
        float rs = rstat[wave][r];
        size_t obase = ((size_t)b * 8192 + q0 + r) * 256 + m * 32;
        ctx[obase + (lane & 15)] = (_Float16)(acc0[i] * rs);
        ctx[obase + 16 + (lane & 15)] = (_Float16)(acc1[i] * rs);
    }
}

// ---------------------------------------------------------------------------
// Residual + LayerNorm: gq = LN(query + mha) * g + b.  One wave per 256-col row.
// ---------------------------------------------------------------------------
__global__ __launch_bounds__(128) void ln_kernel(const float* __restrict__ query,
                                                 const float* __restrict__ mha,
                                                 const float* __restrict__ g,
                                                 const float* __restrict__ bt,
                                                 float* __restrict__ gq) {
    int lane = threadIdx.x & 31;
    int wave = threadIdx.x >> 5;
    int row = blockIdx.x * 4 + wave;
    const float* qp = query + (size_t)row * 256;
    const float* mp = mha + (size_t)row * 256;
    float x[8];
    float s = 0.f;
#pragma unroll
    for (int i = 0; i < 8; ++i) {
        int c = lane * 8 + i;
        x[i] = qp[c] + mp[c];
        s += x[i];
    }
#pragma unroll
    for (int mask = 1; mask < 32; mask <<= 1) s += __shfl_xor(s, mask);
    float mu = s * (1.0f / 256.0f);
    float v = 0.f;
#pragma unroll
    for (int i = 0; i < 8; ++i) { float d = x[i] - mu; v += d * d; }
#pragma unroll
    for (int mask = 1; mask < 32; mask <<= 1) v += __shfl_xor(v, mask);
    float rstd = rsqrtf(v * (1.0f / 256.0f) + 1e-5f);
#pragma unroll
    for (int i = 0; i < 8; ++i) {
        int c = lane * 8 + i;
        gq[(size_t)row * 256 + c] = (x[i] - mu) * rstd * g[c] + bt[c];
    }
}

// ---------------------------------------------------------------------------
// value (B,C,H,W) f32 -> channel-last f16 (b, m, y, x, ch) for coalesced gathers
// ---------------------------------------------------------------------------
__global__ __launch_bounds__(256) void perm_kernel(const float* __restrict__ val,
                                                   _Float16* __restrict__ vp) {
    size_t idx = (size_t)blockIdx.x * 256 + threadIdx.x;  // 16,777,216
    int ch = (int)(idx & 31);
    size_t t = idx >> 5;            // bm*16384 + yx
    int yx = (int)(t & 16383);
    int bm = (int)(t >> 14);
    int b = bm >> 3, m = bm & 7;
    vp[idx] = (_Float16)val[((size_t)(b * 256 + m * 32 + ch)) * 16384 + yx];
}

// ---------------------------------------------------------------------------
// Deformable bilinear sampling + weighted sum over P=4 points.
// One wave = one (b, q, m); lane = channel; wave-uniform coordinates so each
// corner gather is a coalesced 32-lane (64B) read of contiguous channels.
// Writes the reference's faithful (ch*8+m)-scrambled feature layout.
// ---------------------------------------------------------------------------
__global__ __launch_bounds__(128) void sample_kernel(const _Float16* __restrict__ vp,
                                                     const float* __restrict__ refp,
                                                     const float* __restrict__ offs,
                                                     const float* __restrict__ aw,
                                                     _Float16* __restrict__ feat) {
    int lane = threadIdx.x & 31;
    int wave = threadIdx.x >> 5;
    int gid = blockIdx.x * 4 + wave;   // 262144 = 32768 * 8
    int m = gid & 7;
    int r = gid >> 3;                  // b*8192 + q
    int b = r >> 13, q = r & 8191;
    int ch = lane;
    float rx = refp[(size_t)r * 2 + 0];
    float ry = refp[(size_t)r * 2 + 1];
    const _Float16* vbase = vp + (size_t)(b * 8 + m) * 16384 * 32;
    float acc = 0.f;
#pragma unroll
    for (int p = 0; p < 4; ++p) {
        float ox = offs[(size_t)r * 64 + m * 8 + p * 2 + 0];
        float oy = offs[(size_t)r * 64 + m * 8 + p * 2 + 1];
        float w = aw[(size_t)r * 32 + m * 4 + p];
        float gx = (rx + ox) * 128.f - 0.5f;
        float gy = (ry + oy) * 128.f - 0.5f;
        float x0f = floorf(gx), y0f = floorf(gy);
        int x0 = (int)x0f, y0 = (int)y0f;
        int x1 = x0 + 1, y1 = y0 + 1;
        float wx1 = gx - x0f, wx0 = 1.f - wx1;
        float wy1 = gy - y0f, wy0 = 1.f - wy1;
        bool vx0 = (x0 >= 0) & (x0 < 128), vx1 = (x1 >= 0) & (x1 < 128);
        bool vy0 = (y0 >= 0) & (y0 < 128), vy1 = (y1 >= 0) & (y1 < 128);
        float v00 = (vx0 & vy0) ? (float)vbase[((size_t)(y0 * 128 + x0)) * 32 + ch] : 0.f;
        float v01 = (vx1 & vy0) ? (float)vbase[((size_t)(y0 * 128 + x1)) * 32 + ch] : 0.f;
        float v10 = (vx0 & vy1) ? (float)vbase[((size_t)(y1 * 128 + x0)) * 32 + ch] : 0.f;
        float v11 = (vx1 & vy1) ? (float)vbase[((size_t)(y1 * 128 + x1)) * 32 + ch] : 0.f;
        acc += w * (v00 * wx0 * wy0 + v01 * wx1 * wy0 + v10 * wx0 * wy1 + v11 * wx1 * wy1);
    }
    // faithful reorder: out[b, (ch*8+m)*32 + (q>>8), q&255]
    size_t fi = ((size_t)b * 8192 + (size_t)(ch * 8 + m) * 32 + (q >> 8)) * 256 + (q & 255);
    feat[fi] = (_Float16)acc;
}

// ---------------------------------------------------------------------------
// Host: pipeline launch.  Workspace map (bytes, 128 MB total with overlays):
//  [0,1M)      tf f16
//  [1M,2M)     f16 weights: ipw16 (+0), mow16 (+512K), outw16 (+768K)
//  [2M,18M)    query f16        -+
//  [18M,34M)   qf (b,m,q,ch)     +-> reused as gq f32 [2M,34M) after attention
//  [34M,35M)   kf f16
//  [35M,36M)   vf f16 (b,m,ch,t)
//  [36M,52M)   ctx f16 -> reused as feat f16
//  [52M,84M)   mha f32
//  [84M,92M)   offs f32
//  [92M,96M)   attw f32
//  [96M,128M)  value_perm f16
// ---------------------------------------------------------------------------
extern "C" void kernel_launch(void* const* d_in, const int* in_sizes, int n_in,
                              void* d_out, int out_size, void* d_ws, size_t ws_size,
                              hipStream_t stream) {
    const float* query = (const float*)d_in[0];
    const float* value = (const float*)d_in[1];
    const float* refp  = (const float*)d_in[2];
    const float* trajp = (const float*)d_in[3];
    const float* ipw   = (const float*)d_in[4];
    const float* ipb   = (const float*)d_in[5];
    const float* mow   = (const float*)d_in[6];
    const float* mob   = (const float*)d_in[7];
    const float* lng   = (const float*)d_in[8];
    const float* lnb   = (const float*)d_in[9];
    const float* tw    = (const float*)d_in[10];
    const float* tb    = (const float*)d_in[11];
    const float* ofw   = (const float*)d_in[12];
    const float* ofb   = (const float*)d_in[13];
    const float* aww   = (const float*)d_in[14];
    const float* awb   = (const float*)d_in[15];
    const float* outw  = (const float*)d_in[16];
    const float* outb  = (const float*)d_in[17];
    float* out = (float*)d_out;

    char* ws = (char*)d_ws;
    const size_t MB = 1ull << 20;
    _Float16* tf16   = (_Float16*)(ws + 0);
    _Float16* ipw16  = (_Float16*)(ws + 1 * MB);
    _Float16* mow16  = (_Float16*)(ws + 1 * MB + 512 * 1024);
    _Float16* outw16 = (_Float16*)(ws + 1 * MB + 768 * 1024);
    _Float16* q16    = (_Float16*)(ws + 2 * MB);
    _Float16* qf     = (_Float16*)(ws + 18 * MB);
    _Float16* kf     = (_Float16*)(ws + 34 * MB);
    _Float16* vf     = (_Float16*)(ws + 35 * MB);
    _Float16* ctx16  = (_Float16*)(ws + 36 * MB);
    float*    mha    = (float*)(ws + 52 * MB);
    float*    gq     = (float*)(ws + 2 * MB);   // overlays q16+qf (dead by then)
    float*    offs   = (float*)(ws + 84 * MB);
    float*    attw   = (float*)(ws + 92 * MB);
    _Float16* vperm  = (_Float16*)(ws + 96 * MB);
    _Float16* feat   = ctx16;                   // overlays ctx (dead by then)

    // 0. f16 pre-conversion of activations & GEMM weights (numerically identical
    //    to in-loop conversion; removes the cvt storm from the hot loops)
    cvt_kernel<<<32768, 256, 0, stream>>>(query, q16, 8388608);
    cvt_kernel<<<768, 256, 0, stream>>>(ipw, ipw16, 196608);
    cvt_kernel<<<256, 256, 0, stream>>>(mow, mow16, 65536);
    cvt_kernel<<<256, 256, 0, stream>>>(outw, outw16, 65536);
    // 1. trajectory projection (B*Nt=2048 rows x 256, K=2)
    traj_kernel<<<2048, 256, 0, stream>>>(trajp, tw, tb, tf16);
    // 2. value -> channel-last f16
    perm_kernel<<<65536, 256, 0, stream>>>(value, vperm);
    // 3. QKV projections (WMMA f16 + TDM weight staging), head-major scatter
    gemm_nt_f16k<4, 256><<<dim3(512, 4), 128, 0, stream>>>(q16, ipw16, ipb, EpiQ{qf});
    gemm_nt_f16k<4, 256><<<dim3(32, 4), 128, 0, stream>>>(tf16, ipw16 + 65536,
                                                          ipb + 256, EpiK{kf});
    gemm_nt_f16k<4, 256><<<dim3(32, 4), 128, 0, stream>>>(tf16, ipw16 + 131072,
                                                          ipb + 512, EpiV{vf});
    // 4. fused guidance cross-attention
    attn_kernel<<<8192, 64, 0, stream>>>(qf, kf, vf, ctx16);
    // 5. MHA output projection
    gemm_nt_f16k<4, 256><<<dim3(512, 4), 128, 0, stream>>>(ctx16, mow16, mob,
                                                           EpiF32{mha, 256});
    // 6. residual + LayerNorm
    ln_kernel<<<8192, 128, 0, stream>>>(query, mha, lng, lnb, gq);
    // 7. sampling offsets (f32 WMMA, coordinate-critical) + fused-softmax weights
    gemm_nt_f32k<4><<<dim3(512, 1), 128, 0, stream>>>(gq, ofw, ofb, 256,
                                                      EpiF32{offs, 64});
    gemm_nt_f32k<2><<<dim3(512, 1), 128, 0, stream>>>(gq, aww, awb, 256,
                                                      EpiAttw{attw});
    // 8. deformable bilinear sampling + weighted sum (+ faithful reorder)
    sample_kernel<<<65536, 128, 0, stream>>>(vperm, refp, offs, attw, feat);
    // 9. final output projection
    gemm_nt_f16k<4, 256><<<dim3(512, 4), 128, 0, stream>>>(feat, outw16, outb,
                                                           EpiF32{out, 256});
}
